// TaskBranchFine_6691559047589
// MI455X (gfx1250) — compile-verified, matrix operand order checked by reading, not run on hardware
//
#include <hip/hip_runtime.h>
#include <hip/hip_bf16.h>

#define NN 20000
#define NE 640000
#define ETOT (NE + NN)
#define NG 256
#define DD 256
#define ED 11
#define NC 170

typedef __attribute__((ext_vector_type(16))) __bf16 v16bf;
typedef __attribute__((ext_vector_type(8)))  float  v8f;
typedef __attribute__((ext_vector_type(4)))  unsigned int v4u;
typedef __attribute__((ext_vector_type(8)))  int    v8i;
typedef __attribute__((ext_vector_type(4)))  int    v4i;

// LDS row stride for the TDM-staged A tile: 256 + 1 pad DWORD (bank-conflict free)
#define ALDS 257

// ---------- helpers ----------
__device__ __forceinline__ unsigned encf(float f) {
    unsigned u = __float_as_uint(f);
    return (u & 0x80000000u) ? ~u : (u | 0x80000000u);
}
__device__ __forceinline__ float decf(unsigned u) {
    return (u & 0x80000000u) ? __uint_as_float(u & 0x7FFFFFFFu) : __uint_as_float(~u);
}
__device__ __forceinline__ int kpos(int t, int half) {
    int i = t >> 1, j = t & 1;
    int base = (i < 4) ? (i * 2 + j) : (16 + (i - 4) * 2 + j);
    return base + half * 8;
}

// ---------- zero ----------
__global__ void zero_f32(float* __restrict__ p, long n) {
    long i = (long)blockIdx.x * blockDim.x + threadIdx.x;
    if (i < n) p[i] = 0.0f;
}

// ---------- mean of edge_attr over edges (block-reduced) ----------
__global__ __launch_bounds__(256)
void mean_ea_accum(const float* __restrict__ EA, float* __restrict__ macc) {
    __shared__ float sh[256];
    float loc[ED];
#pragma unroll
    for (int k = 0; k < ED; ++k) loc[k] = 0.0f;
    for (int e = blockIdx.x * blockDim.x + threadIdx.x; e < NE;
         e += gridDim.x * blockDim.x) {
        const float* r = EA + (long)e * ED;
#pragma unroll
        for (int k = 0; k < ED; ++k) loc[k] += r[k];
    }
    for (int k = 0; k < ED; ++k) {
        sh[threadIdx.x] = loc[k];
        __syncthreads();
        for (int s = 128; s > 0; s >>= 1) {
            if (threadIdx.x < s) sh[threadIdx.x] += sh[threadIdx.x + s];
            __syncthreads();
        }
        if (threadIdx.x == 0) atomicAdd(&macc[k], sh[0]);
        __syncthreads();
    }
}

// ---------- per-graph node counts ----------
__global__ void count_nodes(const int* __restrict__ batch, float* __restrict__ cnt) {
    int n = blockIdx.x * blockDim.x + threadIdx.x;
    if (n < NN) atomicAdd(&cnt[batch[n]], 1.0f);
}

// ---------- WMMA GEMM with TDM-staged A tile ----------
// Block = one 16-row m-tile (1250 blocks). Wave0 DMAs A[16x256] f32 into LDS
// (padded: row stride 257 DWORDs). Each of 8 waves owns two 16-col n-tiles for
// BOTH weight matrices -> 4 accumulators, 4 WMMAs per K-step sharing one A frag.
__global__ __launch_bounds__(256)
void gemm_xlxr_wmma(const float* __restrict__ X,
                    const float* __restrict__ Wl, const float* __restrict__ bl,
                    const float* __restrict__ Wr, const float* __restrict__ br,
                    float* __restrict__ XL, float* __restrict__ XR) {
    __shared__ float shA[16 * ALDS];            // 16 rows x (256+1) f32
    const int wave = threadIdx.x >> 5;
    const int lane = threadIdx.x & 31;
    const int half = lane >> 4;
    const int lmod = lane & 15;
    const int row0 = blockIdx.x * 16;

    if (wave == 0) {
        // Tensor DMA descriptor: 2D tensor [20000 x 256] f32, tile [16 x 256]
        // starting at row row0; LDS pad: +1 DWORD every 256 DWORDs (interval
        // code 7 = 256, amount code 0 = 1) -> LDS row stride 257.
        unsigned long long ga = (unsigned long long)(const void*)(X + (long)row0 * DD);
        v4u g0;
        g0[0] = 1u;                                    // count=1 (valid user D#)
        g0[1] = 0u;                                    // lds_addr = 0 (shA at LDS base)
        g0[2] = (unsigned)(ga & 0xFFFFFFFFu);          // global_addr[31:0]
        g0[3] = (unsigned)((ga >> 32) & 0x01FFFFFFu)   // global_addr[56:32]
                | 0x80000000u;                         // type = 2 ("image")
        v8i g1;
        g1[0] = (int)(0x20000u                          // data_size = 4 bytes
                | (1u << 20)                            // pad_enable
                | (7u << 22)                            // pad_interval = 256 DWORDs
                | (0u << 25));                          // pad_amount = 1 DWORD
        g1[1] = (int)(256u << 16);                      // tensor_dim0 = 256  [79:48]
        g1[2] = (int)(20000u << 16);                    // tensor_dim1 = 20000 [111:80]
        g1[3] = (int)(256u << 16);                      // tile_dim0 = 256 [127:112]
        g1[4] = 16;                                     // tile_dim1 = 16, tile_dim2 = 0
        g1[5] = 256;                                    // tensor_dim0_stride = 256
        g1[6] = 0;
        g1[7] = 0;
        v4i gz4 = {};
        v8i gz8 = {};
        __builtin_amdgcn_tensor_load_to_lds(g0, g1, gz4, gz4, gz8, 0);
        __builtin_amdgcn_s_wait_tensorcnt(0);
    }
    __syncthreads();
    // Escape shA's address: the TDM wrote LDS behind the compiler's back (the
    // descriptor carries the LDS address as a raw integer), so without this
    // the reads below are folded as "uninitialized" and the ds_loads vanish.
    asm volatile("" : : "v"((void*)shA) : "memory");

    const int c0 = wave * 32 + lmod;                    // n-tile pair columns
    const int c1 = c0 + 16;
    v8f aL0 = {}, aR0 = {}, aL1 = {}, aR1 = {};
    for (int k0 = 0; k0 < DD; k0 += 32) {
        v16bf a, b0L, b0R, b1L, b1R;
#pragma unroll
        for (int t = 0; t < 16; ++t) {
            int k = kpos(t, half) + k0;
            a[t] = (__bf16)shA[lmod * ALDS + k];
        }
#pragma unroll
        for (int t = 0; t < 16; ++t) {
            int k = kpos(t, half) + k0;
            b0L[t] = (__bf16)Wl[k * DD + c0];
            b0R[t] = (__bf16)Wr[k * DD + c0];
            b1L[t] = (__bf16)Wl[k * DD + c1];
            b1R[t] = (__bf16)Wr[k * DD + c1];
        }
        aL0 = __builtin_amdgcn_wmma_f32_16x16x32_bf16(false, a, false, b0L, (short)0, aL0, false, false);
        aR0 = __builtin_amdgcn_wmma_f32_16x16x32_bf16(false, a, false, b0R, (short)0, aR0, false, false);
        aL1 = __builtin_amdgcn_wmma_f32_16x16x32_bf16(false, a, false, b1L, (short)0, aL1, false, false);
        aR1 = __builtin_amdgcn_wmma_f32_16x16x32_bf16(false, a, false, b1R, (short)0, aR1, false, false);
    }
    const float bl0 = bl[c0], bl1 = bl[c1];
    const float br0 = br[c0], br1 = br[c1];
#pragma unroll
    for (int r = 0; r < 8; ++r) {
        int m = r + half * 8;
        long o = (long)(row0 + m) * DD;
        XL[o + c0] = aL0[r] + bl0;
        XL[o + c1] = aL1[r] + bl1;
        XR[o + c0] = aR0[r] + br0;
        XR[o + c1] = aR1[r] + br1;
    }
}

// ---------- pass 1: logits + segment max (one wave per edge) ----------
__global__ __launch_bounds__(256)
void edge_logits(const float* __restrict__ XL, const float* __restrict__ XR,
                 const float* __restrict__ We, const float* __restrict__ be,
                 const float* __restrict__ att,
                 const int* __restrict__ srcA, const int* __restrict__ dstA,
                 const float* __restrict__ EA, const float* __restrict__ macc,
                 float* __restrict__ logits, unsigned* __restrict__ mkey) {
    int e = blockIdx.x * 8 + (threadIdx.x >> 5);
    if (e >= ETOT) return;
    const int lane = threadIdx.x & 31;
    int s, d;
    float ea[ED];
    if (e < NE) {
        s = srcA[e]; d = dstA[e];
        const float* r = EA + (long)e * ED;
#pragma unroll
        for (int k = 0; k < ED; ++k) ea[k] = r[k];
    } else {
        s = d = e - NE;
#pragma unroll
        for (int k = 0; k < ED; ++k) ea[k] = macc[k] * (1.0f / (float)NE);
    }
    float acc = 0.0f;
#pragma unroll
    for (int t = 0; t < 8; ++t) {
        int dd = lane + t * 32;
        float ee = be[dd];
#pragma unroll
        for (int k = 0; k < ED; ++k) ee += ea[k] * We[k * DD + dd];
        float v = XL[(long)s * DD + dd] + XR[(long)d * DD + dd] + ee;
        v = (v > 0.0f) ? v : 0.2f * v;          // leaky_relu 0.2
        acc += v * att[dd];
    }
    for (int off = 16; off > 0; off >>= 1) acc += __shfl_down(acc, off, 32);
    if (lane == 0) {
        logits[e] = acc;
        atomicMax(&mkey[d], encf(acc));
    }
}

// ---------- pass 2: exp + segment sum ----------
__global__ void edge_exp(const float* __restrict__ logits,
                         const int* __restrict__ dstA,
                         const unsigned* __restrict__ mkey,
                         float* __restrict__ exb, float* __restrict__ z) {
    int e = blockIdx.x * blockDim.x + threadIdx.x;
    if (e >= ETOT) return;
    int d = (e < NE) ? dstA[e] : (e - NE);
    float ex = __expf(logits[e] - decf(mkey[d]));
    exb[e] = ex;
    atomicAdd(&z[d], ex);
}

// ---------- pass 3: alpha-weighted scatter of xl[s] (one wave per edge) ----------
__global__ __launch_bounds__(256)
void edge_scatter(const float* __restrict__ exb, const float* __restrict__ z,
                  const float* __restrict__ XL,
                  const int* __restrict__ srcA, const int* __restrict__ dstA,
                  float* __restrict__ aggr) {
    int e = blockIdx.x * 8 + (threadIdx.x >> 5);
    if (e >= ETOT) return;
    const int lane = threadIdx.x & 31;
    int s, d;
    if (e < NE) { s = srcA[e]; d = dstA[e]; } else { s = d = e - NE; }
    float alpha = exb[e] / z[d];
#pragma unroll
    for (int t = 0; t < 8; ++t) {
        int dd = lane + t * 32;
        atomicAdd(&aggr[(long)d * DD + dd], XL[(long)s * DD + dd] * alpha);
    }
}

// ---------- residual + outer leaky + fused pooling ----------
__global__ void node_finalize(const float* __restrict__ Xin,
                              const float* __restrict__ aggr,
                              const float* __restrict__ bias,
                              const int* __restrict__ batch,
                              float* __restrict__ Hout,
                              float* __restrict__ psum, unsigned* __restrict__ pmax) {
    long idx = (long)blockIdx.x * blockDim.x + threadIdx.x;
    if (idx >= (long)NN * DD) return;
    int n = (int)(idx >> 8), dd = (int)(idx & 255);
    float h = Xin[idx] + aggr[idx] + bias[dd];
    h = (h > 0.0f) ? h : 0.01f * h;             // outer leaky 0.01
    Hout[idx] = h;
    int g = batch[n];
    atomicAdd(&psum[g * DD + dd], h);
    atomicMax(&pmax[g * DD + dd], encf(h));
}

// ---------- g = x1 + x2 (mean||max pools) ----------
__global__ void build_g(const float* __restrict__ psum1, const unsigned* __restrict__ pmax1,
                        const float* __restrict__ psum2, const unsigned* __restrict__ pmax2,
                        const float* __restrict__ cnt, float* __restrict__ g) {
    int idx = blockIdx.x * blockDim.x + threadIdx.x;
    if (idx >= NG * DD) return;
    int gi = idx >> 8, dd = idx & 255;
    float c = cnt[gi];
    g[gi * 512 + dd]       = (psum1[idx] + psum2[idx]) / c;
    g[gi * 512 + 256 + dd] = decf(pmax1[idx]) + decf(pmax2[idx]);
}

// ---------- final MLP: [256,512]@[512,256] -> lrelu -> @[256,170] ----------
__global__ __launch_bounds__(256)
void mlp(const float* __restrict__ G,
         const float* __restrict__ W1, const float* __restrict__ b1,
         const float* __restrict__ W2, const float* __restrict__ b2,
         float* __restrict__ out) {
    __shared__ float gr[512];
    __shared__ float hid[256];
    const int gi = blockIdx.x, t = threadIdx.x;
    gr[t]       = G[gi * 512 + t];
    gr[t + 256] = G[gi * 512 + 256 + t];
    __syncthreads();
    float acc = b1[t];
    for (int k = 0; k < 512; ++k) acc += gr[k] * W1[k * DD + t];
    hid[t] = (acc > 0.0f) ? acc : 0.01f * acc;
    __syncthreads();
    if (t < NC) {
        float o = b2[t];
        for (int k = 0; k < DD; ++k) o += hid[k] * W2[k * NC + t];
        out[gi * NC + t] = o;
    }
}

// ---------- workspace layout (in 4-byte units) ----------
#define O_XL    0L
#define O_XR    (O_XL + (long)NN * DD)
#define O_H1    (O_XR + (long)NN * DD)
#define O_AGGR  (O_H1 + (long)NN * DD)     // [aggr | mkey | z] zeroed per layer
#define O_MKEY  (O_AGGR + (long)NN * DD)
#define O_Z     (O_MKEY + NN)
#define O_LOG   (O_Z + NN)
#define O_EXB   (O_LOG + ETOT)
#define O_PSUM1 (O_EXB + ETOT)             // [psum1|pmax1|psum2|pmax2|cnt|macc] zeroed once
#define O_PMAX1 (O_PSUM1 + NG * DD)
#define O_PSUM2 (O_PMAX1 + NG * DD)
#define O_PMAX2 (O_PSUM2 + NG * DD)
#define O_CNT   (O_PMAX2 + NG * DD)
#define O_MACC  (O_CNT + NG)
#define O_GB    (O_MACC + 16)

extern "C" void kernel_launch(void* const* d_in, const int* in_sizes, int n_in,
                              void* d_out, int out_size, void* d_ws, size_t ws_size,
                              hipStream_t stream) {
    const float* x    = (const float*)d_in[0];
    const int*   ei   = (const int*)d_in[1];
    const float* ea   = (const float*)d_in[2];
    const int*   bat  = (const int*)d_in[3];
    const float* c1_Wl = (const float*)d_in[5];
    const float* c1_bl = (const float*)d_in[6];
    const float* c1_Wr = (const float*)d_in[7];
    const float* c1_br = (const float*)d_in[8];
    const float* c1_We = (const float*)d_in[9];
    const float* c1_be = (const float*)d_in[10];
    const float* c1_att = (const float*)d_in[11];
    const float* c1_bias = (const float*)d_in[12];
    const float* c2_Wl = (const float*)d_in[13];
    const float* c2_bl = (const float*)d_in[14];
    const float* c2_Wr = (const float*)d_in[15];
    const float* c2_br = (const float*)d_in[16];
    const float* c2_We = (const float*)d_in[17];
    const float* c2_be = (const float*)d_in[18];
    const float* c2_att = (const float*)d_in[19];
    const float* c2_bias = (const float*)d_in[20];
    const float* W1 = (const float*)d_in[21];
    const float* b1 = (const float*)d_in[22];
    const float* W2 = (const float*)d_in[23];
    const float* b2 = (const float*)d_in[24];

    float* ws = (float*)d_ws;
    const int* srcA = ei;
    const int* dstA = ei + NE;

    float*    XL   = ws + O_XL;
    float*    XR   = ws + O_XR;
    float*    H1   = ws + O_H1;
    float*    AGGR = ws + O_AGGR;
    unsigned* MKEY = (unsigned*)(ws + O_MKEY);
    float*    Z    = ws + O_Z;
    float*    LOG  = ws + O_LOG;
    float*    EXB  = ws + O_EXB;
    float*    PS1  = ws + O_PSUM1;
    unsigned* PM1  = (unsigned*)(ws + O_PMAX1);
    float*    PS2  = ws + O_PSUM2;
    unsigned* PM2  = (unsigned*)(ws + O_PMAX2);
    float*    CNT  = ws + O_CNT;
    float*    MACC = ws + O_MACC;
    float*    GB   = ws + O_GB;

    const int TB = 256;
    const long n_pool = 4L * NG * DD + NG + 16;       // psum1..macc
    const long n_lay  = (long)NN * DD + NN + NN;      // aggr + mkey + z
    const int gEdgeW = (ETOT + 7) / 8;                // wave-per-edge kernels
    const int gEdgeT = (ETOT + TB - 1) / TB;
    const int gNode  = ((long)NN * DD + TB - 1) / TB;
    const int gGemm  = NN / 16;                       // block per 16-row strip

    // one-time accumulator init (per call; harness does not re-zero ws)
    zero_f32<<<(int)((n_pool + TB - 1) / TB), TB, 0, stream>>>(PS1, n_pool);
    count_nodes<<<(NN + TB - 1) / TB, TB, 0, stream>>>(bat, CNT);
    mean_ea_accum<<<256, TB, 0, stream>>>(ea, MACC);

    // ---- layer 1 ----
    zero_f32<<<(int)((n_lay + TB - 1) / TB), TB, 0, stream>>>(AGGR, n_lay);
    gemm_xlxr_wmma<<<gGemm, TB, 0, stream>>>(x, c1_Wl, c1_bl, c1_Wr, c1_br, XL, XR);
    edge_logits<<<gEdgeW, TB, 0, stream>>>(XL, XR, c1_We, c1_be, c1_att,
                                           srcA, dstA, ea, MACC, LOG, MKEY);
    edge_exp<<<gEdgeT, TB, 0, stream>>>(LOG, dstA, MKEY, EXB, Z);
    edge_scatter<<<gEdgeW, TB, 0, stream>>>(EXB, Z, XL, srcA, dstA, AGGR);
    node_finalize<<<gNode, TB, 0, stream>>>(x, AGGR, c1_bias, bat, H1, PS1, PM1);

    // ---- layer 2 ----
    zero_f32<<<(int)((n_lay + TB - 1) / TB), TB, 0, stream>>>(AGGR, n_lay);
    gemm_xlxr_wmma<<<gGemm, TB, 0, stream>>>(H1, c2_Wl, c2_bl, c2_Wr, c2_br, XL, XR);
    edge_logits<<<gEdgeW, TB, 0, stream>>>(XL, XR, c2_We, c2_be, c2_att,
                                           srcA, dstA, ea, MACC, LOG, MKEY);
    edge_exp<<<gEdgeT, TB, 0, stream>>>(LOG, dstA, MKEY, EXB, Z);
    edge_scatter<<<gEdgeW, TB, 0, stream>>>(EXB, Z, XL, srcA, dstA, AGGR);
    node_finalize<<<gNode, TB, 0, stream>>>(H1, AGGR, c2_bias, bat, XL /*h2*/, PS2, PM2);

    // ---- readout ----
    build_g<<<(NG * DD + TB - 1) / TB, TB, 0, stream>>>(PS1, PM1, PS2, PM2, CNT, GB);
    mlp<<<NG, TB, 0, stream>>>(GB, W1, b1, W2, b2, (float*)d_out);
}
